// GeometricRound_8186207666941
// MI455X (gfx1250) — compile-verified, matrix operand order checked by reading, not run on hardware
//
#include <hip/hip_runtime.h>

typedef __attribute__((ext_vector_type(16))) _Float16 v16h;
typedef __attribute__((ext_vector_type(8)))  _Float16 v8h;
typedef __attribute__((ext_vector_type(4)))  _Float16 v4h;
typedef __attribute__((ext_vector_type(8)))  float    v8f;

#define LDS_FENCE()  asm volatile("s_wait_dscnt 0" ::: "memory")
#define ASYNC_WAIT() asm volatile("s_wait_asynccnt 0" ::: "memory")

__device__ __forceinline__ float sigmoidf_(float v) { return 1.0f / (1.0f + __expf(-v)); }

// One 16-byte async copy global -> LDS (CDNA5 ASYNCcnt path, GV addressing form).
__device__ __forceinline__ void async_copy16(const _Float16* gsrc, _Float16* ldst) {
  asm volatile("global_load_async_to_lds_b128 %0, %1, off"
               :: "v"((unsigned)(unsigned long long)(uintptr_t)ldst),
                  "v"((unsigned long long)(uintptr_t)gsrc)
               : "memory");
}

// Build a v16h WMMA fragment from two contiguous 8-half (16B) chunks:
// elements 0..7 = K base..base+7, elements 8..15 = K base+16..base+23 (per-half-wave ISA layout).
#define FRAG16(lo_, hi_) __builtin_shufflevector((lo_), (hi_), 0,1,2,3,4,5,6,7,8,9,10,11,12,13,14,15)

#define LOADB(Wp, stride_, row_, k0_)                                      \
  FRAG16(*(const v8h*)((Wp) + (size_t)(row_) * (stride_) + (k0_)),         \
         *(const v8h*)((Wp) + (size_t)(row_) * (stride_) + (k0_) + 16))

// ---------------------------------------------------------------------------
// prep: fold Cayley signs with sigmoid(interaction_weights) -> coeff[8][8],
// sigmoid(geo_gate), and convert the three FFN weight matrices to f16.
// ---------------------------------------------------------------------------
__global__ void prep_kernel(const float* __restrict__ iw, const float* __restrict__ gg,
                            const float* __restrict__ wg, const float* __restrict__ wu,
                            const float* __restrict__ wd,
                            float* __restrict__ wsf,
                            _Float16* __restrict__ wg16, _Float16* __restrict__ wu16,
                            _Float16* __restrict__ wd16) {
  const int idx = blockIdx.x * blockDim.x + threadIdx.x;
  if (idx < 64) {
    const int k = idx >> 3, i = idx & 7, j = i ^ k;
    int a = i >> 1, total = 0;
    while (a) { total += __popc(a & j); a >>= 1; }
    const float s = (total & 1) ? -1.0f : 1.0f;
    wsf[idx] = s * sigmoidf_(iw[i * 8 + j]);
  }
  if (idx == 64) wsf[64] = sigmoidf_(gg[0]);
  const int stride = gridDim.x * blockDim.x;
  for (int e = idx; e < 512 * 128; e += stride) {
    wg16[e] = (_Float16)wg[e];
    wu16[e] = (_Float16)wu[e];
    wd16[e] = (_Float16)wd[e];
  }
}

// ---------------------------------------------------------------------------
// Fused: geometric product + gate mix + per-blade LayerNorm + SwiGLU FFN + residual.
// 256 threads = 8 wave32; each wave owns a 16-row tile (2 tokens x 8 blades).
// Weight chunks are staged block-wide into LDS via async-to-LDS copies so each
// weight byte is read from L2 once per 128 rows instead of once per 16.
// ---------------------------------------------------------------------------
__global__ void __launch_bounds__(256) geo_ffn_kernel(
    const float* __restrict__ x,
    const float* __restrict__ ln_w, const float* __restrict__ ln_b,
    const float* __restrict__ wsf,
    const _Float16* __restrict__ wg16, const _Float16* __restrict__ wu16,
    const _Float16* __restrict__ wd16,
    float* __restrict__ out) {
  __shared__ float coeff_s[65];
  __shared__ __align__(16) _Float16 wgs[32 * 128];          // 8 KB: gate rows f0..f0+31
  __shared__ __align__(16) _Float16 wus[32 * 128];          // 8 KB: up   rows f0..f0+31
  __shared__ __align__(16) _Float16 wds[128 * 32];          // 8 KB: down cols f0..f0+31 (row-major [d][32])
  __shared__ __align__(16) unsigned char pwb[8 * 4096];     // 4 KB scratch per wave

  const int tid  = threadIdx.x;
  const int lane = tid & 31;
  const int wave = tid >> 5;
  if (tid < 65) coeff_s[tid] = wsf[tid];
  __syncthreads();

  const float gate = coeff_s[64];
  const float omg  = 1.0f - gate;

  const int tile = blockIdx.x * 8 + wave;  // 16-row (token,blade) tile
  const int m16  = lane & 15;              // row (A) / col (B,C,D) within tile
  const int kb   = (lane >> 4) * 8;        // K-base for fragments; also M-offset for C/D

  unsigned char* wb = pwb + wave * 4096;

  const float4 lnw = ((const float4*)ln_w)[lane];
  const float4 lnb = ((const float4*)ln_b)[lane];

  // ---- prologue: geo + gate-mix + LayerNorm -> f16 normed[16][128] in LDS ----
  for (int tok = 0; tok < 2; ++tok) {
    const float* xp = x + ((size_t)(tile * 2 + tok)) * 1024 + lane * 4;
    float4 xi[8];
#pragma unroll
    for (int i = 0; i < 8; ++i) xi[i] = *(const float4*)(xp + i * 128);
#pragma unroll
    for (int k = 0; k < 8; ++k) {
      float4 geo = make_float4(0.f, 0.f, 0.f, 0.f);
#pragma unroll
      for (int i = 0; i < 8; ++i) {
        const float  c = coeff_s[k * 8 + i];
        const float4 a = xi[i];
        const float4 b = xi[i ^ k];
        geo.x += c * a.x * b.x;
        geo.y += c * a.y * b.y;
        geo.z += c * a.z * b.z;
        geo.w += c * a.w * b.w;
      }
      float4 mx;
      mx.x = gate * geo.x + omg * xi[k].x;
      mx.y = gate * geo.y + omg * xi[k].y;
      mx.z = gate * geo.z + omg * xi[k].z;
      mx.w = gate * geo.w + omg * xi[k].w;
      float s  = mx.x + mx.y + mx.z + mx.w;
      float s2 = mx.x * mx.x + mx.y * mx.y + mx.z * mx.z + mx.w * mx.w;
#pragma unroll
      for (int off = 16; off > 0; off >>= 1) {
        s  += __shfl_xor(s,  off, 32);
        s2 += __shfl_xor(s2, off, 32);
      }
      const float mu  = s * (1.0f / 128.0f);
      const float var = s2 * (1.0f / 128.0f) - mu * mu;
      const float inv = rsqrtf(var + 1e-5f);
      v4h nv;
      nv[0] = (_Float16)((mx.x - mu) * inv * lnw.x + lnb.x);
      nv[1] = (_Float16)((mx.y - mu) * inv * lnw.y + lnb.y);
      nv[2] = (_Float16)((mx.z - mu) * inv * lnw.z + lnb.z);
      nv[3] = (_Float16)((mx.w - mu) * inv * lnw.w + lnb.w);
      *(v4h*)(wb + (tok * 8 + k) * 256 + lane * 8) = nv;
    }
  }
  __syncthreads();

  // ---- A fragments: normed 16x128, K-chunks of 32 ----
  v16h afrag[4];
#pragma unroll
  for (int c = 0; c < 4; ++c) {
    const v8h lo = *(const v8h*)(wb + m16 * 256 + (c * 32 + kb) * 2);
    const v8h hi = *(const v8h*)(wb + m16 * 256 + (c * 32 + kb + 16) * 2);
    afrag[c] = FRAG16(lo, hi);
  }
  LDS_FENCE();  // normed region is dead after this; wb reused for h staging

  const v8f vzero = {0, 0, 0, 0, 0, 0, 0, 0};
  v8f dacc[8];
#pragma unroll
  for (int dt = 0; dt < 8; ++dt) dacc[dt] = vzero;

  // ---- FFN: 16 chunks of 32 hidden units ----
  for (int fc = 0; fc < 16; ++fc) {
    const int f0 = fc * 32;

    __syncthreads();  // previous iteration's consumers are done with staging buffers
    // Cooperative async staging: 1536 x 16B chunks, 6 per thread.
#pragma unroll
    for (int i = 0; i < 6; ++i) {
      const int c = tid + 256 * i;
      if (c < 512) {
        async_copy16(wg16 + (size_t)f0 * 128 + c * 8, wgs + c * 8);
      } else if (c < 1024) {
        const int c2 = c - 512;
        async_copy16(wu16 + (size_t)f0 * 128 + c2 * 8, wus + c2 * 8);
      } else {
        const int c2  = c - 1024;
        const int row = c2 >> 2, part = c2 & 3;
        async_copy16(wd16 + (size_t)row * 512 + f0 + part * 8, wds + c2 * 8);
      }
    }
    ASYNC_WAIT();
    __syncthreads();  // staged weights visible block-wide

    v8f g0 = vzero, g1 = vzero, u0 = vzero, u1 = vzero;
#pragma unroll
    for (int c = 0; c < 4; ++c) {
      const int k0 = c * 32 + kb;
      const v16h bg0 = LOADB(wgs, 128, m16,      k0);
      const v16h bg1 = LOADB(wgs, 128, 16 + m16, k0);
      const v16h bu0 = LOADB(wus, 128, m16,      k0);
      const v16h bu1 = LOADB(wus, 128, 16 + m16, k0);
      g0 = __builtin_amdgcn_wmma_f32_16x16x32_f16(false, afrag[c], false, bg0, (short)0, g0, false, false);
      g1 = __builtin_amdgcn_wmma_f32_16x16x32_f16(false, afrag[c], false, bg1, (short)0, g1, false, false);
      u0 = __builtin_amdgcn_wmma_f32_16x16x32_f16(false, afrag[c], false, bu0, (short)0, u0, false, false);
      u1 = __builtin_amdgcn_wmma_f32_16x16x32_f16(false, afrag[c], false, bu1, (short)0, u1, false, false);
    }
    // h = silu(g)*u -> per-wave LDS chunk [16 rows][32 halves]
#pragma unroll
    for (int r = 0; r < 8; ++r) {
      const int m = r + kb;  // C/D row for this vgpr slot
      const float h0 = (g0[r] * sigmoidf_(g0[r])) * u0[r];
      const float h1 = (g1[r] * sigmoidf_(g1[r])) * u1[r];
      *(_Float16*)(wb + m * 64 + m16 * 2)        = (_Float16)h0;
      *(_Float16*)(wb + m * 64 + (16 + m16) * 2) = (_Float16)h1;
    }
    LDS_FENCE();
    // down-projection A fragment from h chunk (K=32 over ffn units)
    const v8h hlo = *(const v8h*)(wb + m16 * 64 + kb * 2);
    const v8h hhi = *(const v8h*)(wb + m16 * 64 + (kb + 16) * 2);
    const v16h ha = FRAG16(hlo, hhi);
#pragma unroll
    for (int dt = 0; dt < 8; ++dt) {
      const v16h bd = LOADB(wds, 32, dt * 16 + m16, kb);
      dacc[dt] = __builtin_amdgcn_wmma_f32_16x16x32_f16(false, ha, false, bd, (short)0, dacc[dt], false, false);
    }
    LDS_FENCE();
  }

  // ---- epilogue: stage f32 result in two 4 KB halves, coalesced residual-add ----
#pragma unroll
  for (int half = 0; half < 2; ++half) {
#pragma unroll
    for (int q = 0; q < 4; ++q) {
      const int dt = half * 4 + q;
#pragma unroll
      for (int r = 0; r < 8; ++r) {
        *(float*)(wb + (r + kb) * 256 + (q * 16 + m16) * 4) = dacc[dt][r];
      }
    }
    LDS_FENCE();
#pragma unroll
    for (int rr = 0; rr < 16; ++rr) {
      const float2 acc = *(const float2*)(wb + rr * 256 + lane * 8);
      const size_t off = ((size_t)tile * 16 + rr) * 128 + half * 64 + lane * 2;
      const float2 xv  = *(const float2*)(x + off);
      float2 o;
      o.x = xv.x + acc.x;
      o.y = xv.y + acc.y;
      *(float2*)(out + off) = o;
    }
    LDS_FENCE();  // half 0 staging fully read before half 1 overwrites
  }
}

// ---------------------------------------------------------------------------
extern "C" void kernel_launch(void* const* d_in, const int* in_sizes, int n_in,
                              void* d_out, int out_size, void* d_ws, size_t ws_size,
                              hipStream_t stream) {
  const float* x   = (const float*)d_in[0];
  const float* iw  = (const float*)d_in[1];
  const float* gg  = (const float*)d_in[2];
  const float* lnw = (const float*)d_in[3];
  const float* lnb = (const float*)d_in[4];
  const float* wg  = (const float*)d_in[5];
  const float* wu  = (const float*)d_in[6];
  const float* wd  = (const float*)d_in[7];
  float* out = (float*)d_out;

  // workspace layout: [0,260) coeff+gate (f32), then f16 weights (512B aligned)
  float*    wsf  = (float*)d_ws;
  _Float16* wg16 = (_Float16*)((char*)d_ws + 512);
  _Float16* wu16 = (_Float16*)((char*)d_ws + 512 + 131072);
  _Float16* wd16 = (_Float16*)((char*)d_ws + 512 + 262144);

  prep_kernel<<<128, 256, 0, stream>>>(iw, gg, wg, wu, wd, wsf, wg16, wu16, wd16);

  const int rows   = in_sizes[0] / 128;  // B*T*8 = 131072
  const int tiles  = rows / 16;          // 8192
  const int blocks = tiles / 8;          // 8 waves/block, one tile per wave
  geo_ffn_kernel<<<blocks, 256, 0, stream>>>(x, lnw, lnb, wsf, wg16, wu16, wd16, out);
}
